// YatSphericalCausalAttention_10187662426974
// MI455X (gfx1250) — matrix operand order repaired and checked by reading, not compiled, not run on hardware
//
#include <hip/hip_runtime.h>
#include <hip/hip_bf16.h>

// Problem constants (from reference): B=2, T=2048, D=1024, H=16, hd=64
#define B_  2
#define T_  2048
#define D_  1024
#define H_  16
#define HD_ 64

typedef __bf16 bf16;
typedef __attribute__((ext_vector_type(16))) __bf16 v16bf;
typedef __attribute__((ext_vector_type(8)))  __bf16 v8bf;
typedef __attribute__((ext_vector_type(8)))  float  v8f;

static __device__ __forceinline__ bf16 f2bf(float f) {
  unsigned u = __builtin_bit_cast(unsigned, f);
  unsigned r = (u + 0x7FFFu + ((u >> 16) & 1u)) >> 16;
  unsigned short h = (unsigned short)r;
  return __builtin_bit_cast(bf16, h);
}

// ---------------------------------------------------------------------------
// fp32 -> bf16 convert
// ---------------------------------------------------------------------------
__global__ void convert_f32_to_bf16(const float* __restrict__ src,
                                    bf16* __restrict__ dst, int n) {
  int i = blockIdx.x * blockDim.x + threadIdx.x;
  if (i < n) dst[i] = f2bf(src[i]);
}

// ---------------------------------------------------------------------------
// C[M,N] (fp32) = A[M,K] (bf16, row-major) @ Bw[N,K]^T (bf16, torch Linear wt)
// One 16x16 WMMA tile per wave, 8 waves/WG -> 64x32 WG tile. K step = 32.
// ---------------------------------------------------------------------------
__global__ void gemm_bf16_wmma(const bf16* __restrict__ A,
                               const bf16* __restrict__ Bw,
                               float* __restrict__ Cf,
                               int M, int N, int K) {
  const int lane = threadIdx.x & 31;
  const int wave = threadIdx.x >> 5;
  const int tilesN = N >> 5;
  const int tM = blockIdx.x / tilesN;
  const int tN = blockIdx.x % tilesN;
  const int row0 = tM * 64 + (wave >> 1) * 16;
  const int col0 = tN * 32 + (wave & 1) * 16;
  const int m  = lane & 15;
  const int hs = lane >> 4;

  const bf16* arow = A  + (size_t)(row0 + m) * K;
  const bf16* bcol = Bw + (size_t)(col0 + m) * K;

  v8f acc = {};
  for (int k0 = 0; k0 < K; k0 += 32) {
    v16bf av, bv;
    {
      const v8bf lo = *(const v8bf*)(arow + k0 + hs * 8);
      const v8bf hi = *(const v8bf*)(arow + k0 + 16 + hs * 8);
#pragma unroll
      for (int i = 0; i < 8; ++i) { av[i] = lo[i]; av[8 + i] = hi[i]; }
    }
    {
      const v8bf lo = *(const v8bf*)(bcol + k0 + hs * 8);
      const v8bf hi = *(const v8bf*)(bcol + k0 + 16 + hs * 8);
#pragma unroll
      for (int i = 0; i < 8; ++i) { bv[i] = lo[i]; bv[8 + i] = hi[i]; }
    }
    acc = __builtin_amdgcn_wmma_f32_16x16x32_bf16(false, av, false, bv,
                                                  (short)0, acc, false, false);
  }

  const int mBase = hs * 8;
#pragma unroll
  for (int i = 0; i < 8; ++i) {
    Cf[(size_t)(row0 + mBase + i) * N + (col0 + m)] = acc[i];
  }
}

// ---------------------------------------------------------------------------
// Split qkv [B*T, 3D] fp32 into:
//   qn, kn : head-major bf16 [B*H, T, 64], L2-normalized along head dim
//   vT     : tile-transposed bf16 [B*H, T/32, 64, 32] so attention P@V
//            B-operands are contiguous b128s.
// ---------------------------------------------------------------------------
__global__ void qkv_normalize_split(const float* __restrict__ qkv,
                                    bf16* __restrict__ qn,
                                    bf16* __restrict__ kn,
                                    bf16* __restrict__ vT) {
  int item = blockIdx.x * blockDim.x + threadIdx.x; // (b, t, h)
  if (item >= B_ * T_ * H_) return;
  int h = item % H_;
  int t = (item / H_) % T_;
  int b = item / (H_ * T_);
  int bh = b * H_ + h;

  const float* base = qkv + (size_t)(b * T_ + t) * (3 * D_);
  const float* qrow = base + h * HD_;
  const float* krow = base + D_ + h * HD_;
  const float* vrow = base + 2 * D_ + h * HD_;
  size_t orow = ((size_t)bh * T_ + t) * HD_;

  float ssq = 0.f, ssk = 0.f;
#pragma unroll 8
  for (int d = 0; d < HD_; ++d) { ssq += qrow[d] * qrow[d]; ssk += krow[d] * krow[d]; }
  float sq = 1.f / fmaxf(sqrtf(ssq), 1e-12f);
  float sk = 1.f / fmaxf(sqrtf(ssk), 1e-12f);

  size_t vtBase = ((size_t)bh * (T_ / 32) + (t >> 5)) * (HD_ * 32) + (t & 31);
#pragma unroll 8
  for (int d = 0; d < HD_; ++d) {
    qn[orow + d] = f2bf(qrow[d] * sq);
    kn[orow + d] = f2bf(krow[d] * sk);
    vT[vtBase + (size_t)d * 32] = f2bf(vrow[d]);
  }
}

// ---------------------------------------------------------------------------
// Causal Yat attention, flash-style. One WG = 8 waves = 128 rows of one (b,h).
// Each wave owns a 16-row strip; loops s-tiles of 32 up to its diagonal.
// K tiles are double-buffered into per-wave LDS via gfx1250 async loads
// (GLOBAL_LOAD_ASYNC_TO_LDS_B128, ASYNCcnt), overlapping HBM/L2 latency with
// the previous tile's WMMA work.
//   S(16x32)  = qn(16x64) @ kn^T        -> 2x2 WMMA, fp32 accum (K from LDS)
//   Kscore    = S^2 / max(2.01-2S, 1e-6), causal mask, per-row sums kept
//   Kscore staged to LDS (bf16), reloaded in WMMA A-layout
//   num(16x64) += Kscore(16x32) @ V(32x64) -> 4 WMMAs, V from vT (contiguous)
// Final: out = num / (rowsum + 1e-6) written bf16 to attn [B,T,D].
// ---------------------------------------------------------------------------
__global__ void yat_attention_wmma(const bf16* __restrict__ qn,
                                   const bf16* __restrict__ kn,
                                   const bf16* __restrict__ vT,
                                   bf16* __restrict__ attn) {
  __shared__ __align__(16) bf16 sCore[8][16][32];   //  8 KB score staging
  __shared__ __align__(16) bf16 sK[8][2][32 * HD_]; // 64 KB K dbl-buffer

  const int lane = threadIdx.x & 31;
  const int wave = threadIdx.x >> 5;
  const int numRT = T_ / 128;
  const int bh = blockIdx.x / numRT;
  const int rt = blockIdx.x % numRT;
  const int h = bh % H_;
  const int b = bh / H_;
  const int i0 = rt * 128 + wave * 16;
  const int m  = lane & 15;
  const int hs = lane >> 4;
  const int mBase = hs * 8;

  const bf16* qbase = qn + (size_t)bh * T_ * HD_;
  const bf16* kbase = kn + (size_t)bh * T_ * HD_;
  const bf16* vtb   = vT + (size_t)bh * (T_ / 32) * (HD_ * 32);

  // Async-copy one 32x64 bf16 K tile (4 KB, contiguous in kn) into LDS.
  // 8 issues/wave, each lane moves 16 B per issue (512 B per instruction).
  auto issueK = [&](int j0, int buf) {
    unsigned dst0 = (unsigned)(size_t)&sK[wave][buf][0] + (unsigned)(lane * 16);
    const bf16* src0 = kbase + (size_t)j0 * HD_ + lane * 8;
#pragma unroll
    for (int q = 0; q < 8; ++q) {
      unsigned dst = dst0 + (unsigned)(q * 512);
      const bf16* src = src0 + q * 256;
      asm volatile("global_load_async_to_lds_b128 %0, %1, off"
                   :: "v"(dst), "v"(src) : "memory");
    }
  };

  // Q operand (A-matrix) for this wave's 16 rows: K=64 -> two 32-chunks.
  v16bf aq[2];
#pragma unroll
  for (int c = 0; c < 2; ++c) {
    const bf16* p = qbase + (size_t)(i0 + m) * HD_ + c * 32;
    v8bf lo = *(const v8bf*)(p + hs * 8);
    v8bf hi = *(const v8bf*)(p + 16 + hs * 8);
#pragma unroll
    for (int i = 0; i < 8; ++i) { aq[c][i] = lo[i]; aq[c][8 + i] = hi[i]; }
  }

  v8f acc[4] = {};                  // numerator, 4 d-tiles of 16
  float dsum[8];
#pragma unroll
  for (int i = 0; i < 8; ++i) dsum[i] = 0.f;

  issueK(0, 0);                     // prime the pipeline
  int it = 0;
  for (int j0 = 0; j0 < i0 + 16; j0 += 32, ++it) {
    const int buf = it & 1;
    const bool more = (j0 + 32) < (i0 + 16);
    if (more) {
      issueK(j0 + 32, buf ^ 1);
      // In-order async completion: <=8 outstanding => current tile landed,
      // only the just-issued next tile may still be in flight.
      asm volatile("s_wait_asynccnt 0x8" ::: "memory");
    } else {
      asm volatile("s_wait_asynccnt 0x0" ::: "memory");
    }

    // ---- scores: two 16-col groups, K operand from LDS (ds_load_b128) ----
    v8f sc[2];
#pragma unroll
    for (int g = 0; g < 2; ++g) {
      v16bf bk[2];
#pragma unroll
      for (int c = 0; c < 2; ++c) {
        const bf16* p = &sK[wave][buf][(g * 16 + m) * HD_ + c * 32 + hs * 8];
        v8bf lo = *(const v8bf*)(p);
        v8bf hi = *(const v8bf*)(p + 16);
#pragma unroll
        for (int i = 0; i < 8; ++i) { bk[c][i] = lo[i]; bk[c][8 + i] = hi[i]; }
      }
      v8f s = {};
      s = __builtin_amdgcn_wmma_f32_16x16x32_bf16(false, aq[0], false, bk[0],
                                                  (short)0, s, false, false);
      s = __builtin_amdgcn_wmma_f32_16x16x32_bf16(false, aq[1], false, bk[1],
                                                  (short)0, s, false, false);
      sc[g] = s;
    }

    // ---- Yat transform + causal mask + row-sum + LDS stash ----
#pragma unroll
    for (int g = 0; g < 2; ++g) {
      const int col = j0 + g * 16 + m;
#pragma unroll
      for (int i = 0; i < 8; ++i) {
        const int row = i0 + mBase + i;
        float s = sc[g][i];
        float ks = 0.f;
        if (col <= row) {
          float den = fmaxf(2.01f - 2.f * s, 1e-6f);
          ks = s * s / den;
        }
        dsum[i] += ks;
        sCore[wave][mBase + i][g * 16 + m] = f2bf(ks);
      }
    }
    // Per-wave LDS ordering: make stores visible before the A-layout reload.
    asm volatile("s_wait_dscnt 0" ::: "memory");

    // ---- reload scores in WMMA A-layout (16x32) ----
    v16bf sA;
    {
      const bf16* p = &sCore[wave][m][0];
      v8bf lo = *(const v8bf*)(p + hs * 8);
      v8bf hi = *(const v8bf*)(p + 16 + hs * 8);
#pragma unroll
      for (int i = 0; i < 8; ++i) { sA[i] = lo[i]; sA[8 + i] = hi[i]; }
    }

    // ---- numerator += Kscore @ V : 4 d-tiles, V from vT (contiguous) ----
    const bf16* vtile = vtb + (size_t)(j0 >> 5) * (HD_ * 32);
#pragma unroll
    for (int j = 0; j < 4; ++j) {
      const int dcol = j * 16 + m;
      const bf16* p = vtile + (size_t)dcol * 32 + hs * 8;
      v16bf bvv;
      v8bf lo = *(const v8bf*)(p);
      v8bf hi = *(const v8bf*)(p + 16);
#pragma unroll
      for (int kk = 0; kk < 8; ++kk) { bvv[kk] = lo[kk]; bvv[8 + kk] = hi[kk]; }
      acc[j] = __builtin_amdgcn_wmma_f32_16x16x32_bf16(false, sA, false, bvv,
                                                       (short)0, acc[j], false, false);
    }
  }

  // Row-sum reduction across the 16 lanes of each half-wave.
#pragma unroll
  for (int i = 0; i < 8; ++i) {
    float v = dsum[i];
    v += __shfl_xor(v, 1, 32);
    v += __shfl_xor(v, 2, 32);
    v += __shfl_xor(v, 4, 32);
    v += __shfl_xor(v, 8, 32);
    dsum[i] = v + 1e-6f;
  }

  // Write attn [B, T, D] bf16 (t-major so GEMM2 reads contiguous rows).
#pragma unroll
  for (int j = 0; j < 4; ++j) {
#pragma unroll
    for (int i = 0; i < 8; ++i) {
      size_t idx = (size_t)(b * T_ + i0 + mBase + i) * D_ + h * HD_ + j * 16 + m;
      attn[idx] = f2bf(acc[j][i] / dsum[i]);
    }
  }
}

// ---------------------------------------------------------------------------
// Host-side orchestration (all launches on `stream`, graph-capture safe)
// ---------------------------------------------------------------------------
extern "C" void kernel_launch(void* const* d_in, const int* in_sizes, int n_in,
                              void* d_out, int out_size, void* d_ws, size_t ws_size,
                              hipStream_t stream) {
  const float* x    = (const float*)d_in[0];
  const float* Wqkv = (const float*)d_in[1];
  const float* Wout = (const float*)d_in[2];
  float* out = (float*)d_out;

  char* ws = (char*)d_ws;
  size_t off = 0;
  auto take = [&](size_t bytes) -> char* {
    char* p = ws + off;
    off += (bytes + 255) & ~(size_t)255;
    return p;
  };

  bf16*  xb    = (bf16*)take((size_t)B_ * T_ * D_ * 2);        //  8 MB
  bf16*  wqkvb = (bf16*)take((size_t)3 * D_ * D_ * 2);         //  6 MB
  bf16*  woutb = (bf16*)take((size_t)D_ * D_ * 2);             //  2 MB
  float* qkvf  = (float*)take((size_t)B_ * T_ * 3 * D_ * 4);   // 48 MB
  bf16*  qn    = (bf16*)take((size_t)B_ * H_ * T_ * HD_ * 2);  //  8 MB
  bf16*  kn    = (bf16*)take((size_t)B_ * H_ * T_ * HD_ * 2);  //  8 MB
  bf16*  vbt   = (bf16*)take((size_t)B_ * H_ * T_ * HD_ * 2);  //  8 MB
  bf16*  attnb = (bf16*)take((size_t)B_ * T_ * D_ * 2);        //  8 MB
  (void)in_sizes; (void)n_in; (void)out_size; (void)ws_size;

  int n;
  n = B_ * T_ * D_;
  convert_f32_to_bf16<<<(n + 255) / 256, 256, 0, stream>>>(x, xb, n);
  n = 3 * D_ * D_;
  convert_f32_to_bf16<<<(n + 255) / 256, 256, 0, stream>>>(Wqkv, wqkvb, n);
  n = D_ * D_;
  convert_f32_to_bf16<<<(n + 255) / 256, 256, 0, stream>>>(Wout, woutb, n);

  const int M = B_ * T_;  // 4096
  // qkv = x @ Wqkv^T  (M=4096, N=3072, K=1024)
  gemm_bf16_wmma<<<(M / 64) * (3 * D_ / 32), 256, 0, stream>>>(
      xb, wqkvb, qkvf, M, 3 * D_, D_);

  qkv_normalize_split<<<(B_ * T_ * H_ + 255) / 256, 256, 0, stream>>>(
      qkvf, qn, kn, vbt);

  yat_attention_wmma<<<B_ * H_ * (T_ / 128), 256, 0, stream>>>(qn, kn, vbt, attnb);

  // out = attn @ Wout^T  (M=4096, N=1024, K=1024), fp32 to d_out
  gemm_bf16_wmma<<<(M / 64) * (D_ / 32), 256, 0, stream>>>(
      attnb, woutb, out, M, D_, D_);
}
